// PointMultiGraspNet_PointNext_90658169683992
// MI455X (gfx1250) — matrix operand order repaired and checked by reading, not compiled, not run on hardware
//
#include <hip/hip_runtime.h>
#include <hip/hip_bf16.h>

typedef __attribute__((ext_vector_type(16))) _Float16 v16h;
typedef __attribute__((ext_vector_type(8)))  float    v8f;

typedef __attribute__((ext_vector_type(4))) unsigned int tdm_v4u;
typedef __attribute__((ext_vector_type(8))) int          tdm_v8i;
typedef __attribute__((ext_vector_type(4))) int          tdm_v4i;

#define B_       128
#define NSAMP_   32
#define CHUNK_B  16

static __device__ __forceinline__ v8f wmma_f16(v16h a, v16h b, v8f c) {
  // D = A(16x32) * B(32x16) + C, f32 accumulate
  return __builtin_amdgcn_wmma_f32_16x16x32_f16(false, a, false, b, (short)0, c, false, false);
}

// A-operand K index for element j of the v16h (16-bit A 16x32 layout)
static __device__ __forceinline__ int a_kidx(int k0, int hi, int j) {
  return k0 + hi * 8 + j + ((j >> 3) << 3);   // j<8 -> j ; j>=8 -> j+8
}

// ---------------------------------------------------------------------------
// Weight packing: fp32 [rows][cols] -> f16 [rowsPad][colsPad], zero padded
// ---------------------------------------------------------------------------
__global__ void k_pack_w(const float* __restrict__ src, _Float16* __restrict__ dst,
                         int rows, int cols, int rowsPad, int colsPad) {
  int t = blockIdx.x * blockDim.x + threadIdx.x;
  int total = rowsPad * colsPad;
  if (t >= total) return;
  int r = t / colsPad, c = t % colsPad;
  float v = (r < rows && c < cols) ? src[(size_t)r * cols + c] : 0.f;
  dst[t] = (_Float16)v;
}

// ---------------------------------------------------------------------------
// Head: points(B,N,4) -> p0(B,N,3) fp32 and f0(B,N,32) f16
// ---------------------------------------------------------------------------
__global__ void k_head(const float* __restrict__ pts, const float* __restrict__ hw,
                       const float* __restrict__ hb, float* __restrict__ p0,
                       _Float16* __restrict__ f0, int N) {
  int t = blockIdx.x * blockDim.x + threadIdx.x;
  if (t >= B_ * N) return;
  const float* q = pts + (size_t)t * 4;
  float x = q[0], y = q[1], z = q[2], e = q[3];
  p0[(size_t)t * 3 + 0] = x; p0[(size_t)t * 3 + 1] = y; p0[(size_t)t * 3 + 2] = z;
  _Float16* fo = f0 + (size_t)t * 32;
#pragma unroll
  for (int c = 0; c < 32; ++c) {
    float v = hb[c] + hw[c * 4 + 0] * x + hw[c * 4 + 1] * y + hw[c * 4 + 2] * z + hw[c * 4 + 3] * e;
    fo[c] = (_Float16)v;
  }
}

// ---------------------------------------------------------------------------
// Farthest point sampling: one block per batch, LDS argmax reduction
// ---------------------------------------------------------------------------
__global__ void k_fps(const float* __restrict__ p, int N, int S, int* __restrict__ idx) {
  int b = blockIdx.x;
  __shared__ float dist[1024];
  __shared__ float rv[256];
  __shared__ int   ri[256];
  __shared__ float cx, cy, cz;
  __shared__ int   far;
  const float* pb = p + (size_t)b * N * 3;
  for (int i = threadIdx.x; i < N; i += blockDim.x) dist[i] = 1e10f;
  if (threadIdx.x == 0) far = 0;
  __syncthreads();
  for (int it = 0; it < S; ++it) {
    if (threadIdx.x == 0) {
      idx[(size_t)b * S + it] = far;
      cx = pb[far * 3 + 0]; cy = pb[far * 3 + 1]; cz = pb[far * 3 + 2];
    }
    __syncthreads();
    float best = -1.f; int bi = 0;
    for (int i = threadIdx.x; i < N; i += blockDim.x) {
      float dx = pb[i * 3 + 0] - cx, dy = pb[i * 3 + 1] - cy, dz = pb[i * 3 + 2] - cz;
      float nd = fminf(dist[i], dx * dx + dy * dy + dz * dz);
      dist[i] = nd;
      if (nd > best) { best = nd; bi = i; }
    }
    rv[threadIdx.x] = best; ri[threadIdx.x] = bi;
    __syncthreads();
    for (int o = 128; o > 0; o >>= 1) {
      if ((int)threadIdx.x < o) {
        float ov = rv[threadIdx.x + o]; int oi = ri[threadIdx.x + o];
        if (ov > rv[threadIdx.x] || (ov == rv[threadIdx.x] && oi < ri[threadIdx.x])) {
          rv[threadIdx.x] = ov; ri[threadIdx.x] = oi;
        }
      }
      __syncthreads();
    }
    if (threadIdx.x == 0) far = ri[0];
    __syncthreads();
  }
}

// ---------------------------------------------------------------------------
// Ball query: first NSAMP_ indices (ascending) with d2 < r2; pad with first/0
// ---------------------------------------------------------------------------
__global__ void k_ball(const float* __restrict__ p, const float* __restrict__ np_,
                       int* __restrict__ gidx, int N, int S, float r2) {
  int t = blockIdx.x * blockDim.x + threadIdx.x;
  if (t >= B_ * S) return;
  int b = t / S;
  const float* pb = p + (size_t)b * N * 3;
  float cx = np_[(size_t)t * 3 + 0], cy = np_[(size_t)t * 3 + 1], cz = np_[(size_t)t * 3 + 2];
  int* out = gidx + (size_t)t * NSAMP_;
  int cnt = 0, first = 0; bool has = false;
  for (int i = 0; i < N && cnt < NSAMP_; ++i) {
    float dx = pb[i * 3 + 0] - cx, dy = pb[i * 3 + 1] - cy, dz = pb[i * 3 + 2] - cz;
    if (dx * dx + dy * dy + dz * dz < r2) {
      if (!has) { first = i; has = true; }
      out[cnt++] = i;
    }
  }
  for (int j = cnt; j < NSAMP_; ++j) out[j] = has ? first : 0;
}

// ---------------------------------------------------------------------------
// Gather centers: new_p(B,S,3) fp32, fi(B,S,ci) f16
// ---------------------------------------------------------------------------
__global__ void k_gather(const float* __restrict__ p, const _Float16* __restrict__ f,
                         const int* __restrict__ idx, float* __restrict__ np_,
                         _Float16* __restrict__ fi, int N, int S, int ci) {
  int t = blockIdx.x * blockDim.x + threadIdx.x;
  if (t >= B_ * S) return;
  int b = t / S;
  int id = idx[t];
  const float* src = p + ((size_t)b * N + id) * 3;
  np_[(size_t)t * 3 + 0] = src[0]; np_[(size_t)t * 3 + 1] = src[1]; np_[(size_t)t * 3 + 2] = src[2];
  const _Float16* fs = f + ((size_t)b * N + id) * ci;
  _Float16* fd = fi + (size_t)t * ci;
  for (int c = 0; c < ci; ++c) fd[c] = fs[c];
}

// ---------------------------------------------------------------------------
// Generic WMMA GEMM, 1x2 column-tile register blocking:
// out[m][n] = A[m][k] * W[n][k]  (+bias) (*gamma+beta) (relu)
// A: f16 [M][Kpad] row-major, W: f16 [Npad][Kpad]. Npad must be multiple of 32.
// ---------------------------------------------------------------------------
__global__ void k_gemm(const _Float16* __restrict__ A, const _Float16* __restrict__ W,
                       const float* __restrict__ bias,
                       const float* __restrict__ gamma, const float* __restrict__ beta,
                       float* __restrict__ outF, _Float16* __restrict__ outH,
                       int M, int Kpad, int Npad, int Nact, int ldout, int relu) {
  int lane = threadIdx.x & 31;
  int wv   = threadIdx.x >> 5;
  int nt0  = (blockIdx.y * 8 + wv) * 2;
  if (nt0 * 16 >= Npad) return;
  int m0 = blockIdx.x * 16;
  int hi = lane >> 4;
  int mr = m0 + (lane & 15);
  int n0 = nt0 * 16 + (lane & 15);
  v8f acc[2] = {};
  const _Float16* ap  = A + (size_t)mr * Kpad + hi * 8;
  const _Float16* wp0 = W + (size_t)n0 * Kpad + hi * 8;
  const _Float16* wp1 = wp0 + (size_t)16 * Kpad;
  for (int k0 = 0; k0 < Kpad; k0 += 32) {
    v16h a, b0, b1;
#pragma unroll
    for (int j = 0; j < 8; ++j) { a[j] = ap[j]; a[j + 8] = ap[16 + j]; }
#pragma unroll
    for (int j = 0; j < 8; ++j) { b0[j] = wp0[j]; b0[j + 8] = wp0[16 + j]; b1[j] = wp1[j]; b1[j + 8] = wp1[16 + j]; }
    __builtin_prefetch((const void*)(ap + 32), 0, 1);
    acc[0] = wmma_f16(a, b0, acc[0]);
    acc[1] = wmma_f16(a, b1, acc[1]);
    ap += 32; wp0 += 32; wp1 += 32;
  }
#pragma unroll
  for (int tile = 0; tile < 2; ++tile) {
    int n = n0 + tile * 16;
#pragma unroll
    for (int i = 0; i < 8; ++i) {
      int m = m0 + i + hi * 8;
      float v = acc[tile][i];
      bool valid = (n < Nact);
      if (valid) {
        if (bias)  v += bias[n];
        if (gamma) v = v * gamma[n] + beta[n];
        if (relu)  v = fmaxf(v, 0.f);
        if (outF) outF[(size_t)m * ldout + n] = v;
      }
      if (outH) outH[(size_t)m * ldout + n] = (_Float16)(valid ? v : 0.f);
    }
  }
}

// ---------------------------------------------------------------------------
// Stage GEMM1 with fused gather (1x2 column tiles): A rows are (b,s,k) features
// feat = [ (p[g]-new_p)/r (3ch) , f[g] (ci ch) , 0-pad ] ; out h1 = relu(h*g1+b1)
// ---------------------------------------------------------------------------
__global__ void k_gemm1_feat(const float* __restrict__ p, const _Float16* __restrict__ f,
                             const float* __restrict__ np_, const int* __restrict__ gidx,
                             const _Float16* __restrict__ W1,
                             const float* __restrict__ g1, const float* __restrict__ b1,
                             _Float16* __restrict__ h1,
                             int bbase, int S, int Nprev, int ci, int cinp, int mid, float invR) {
  int lane = threadIdx.x & 31;
  int wv   = threadIdx.x >> 5;
  int nt0  = (blockIdx.y * 8 + wv) * 2;
  if (nt0 * 16 >= mid) return;
  int m0 = blockIdx.x * 16;
  int hi = lane >> 4;
  int r  = m0 + (lane & 15);              // chunk-local row
  int bl = r / (S * 32);
  int rem = r - bl * S * 32;
  int b  = bbase + bl;
  int s  = rem >> 5;
  int k  = rem & 31;
  int g  = gidx[(((size_t)b * S) + s) * 32 + k];
  const float* pg = p   + ((size_t)b * Nprev + g) * 3;
  const float* nc = np_ + ((size_t)b * S + s) * 3;
  const _Float16* fg = f + ((size_t)b * Nprev + g) * ci;
  float dp0 = (pg[0] - nc[0]) * invR;
  float dp1 = (pg[1] - nc[1]) * invR;
  float dp2 = (pg[2] - nc[2]) * invR;
  int n0 = nt0 * 16 + (lane & 15);
  const _Float16* wp0 = W1 + (size_t)n0 * cinp + hi * 8;
  const _Float16* wp1 = wp0 + (size_t)16 * cinp;
  v8f acc[2] = {};
  for (int k0 = 0; k0 < cinp; k0 += 32) {
    v16h a, b0, b1;
#pragma unroll
    for (int j = 0; j < 16; ++j) {
      int c = a_kidx(k0, hi, j);
      _Float16 v;
      if (c == 0)      v = (_Float16)dp0;
      else if (c == 1) v = (_Float16)dp1;
      else if (c == 2) v = (_Float16)dp2;
      else if (c < 3 + ci) v = fg[c - 3];
      else v = (_Float16)0.f;
      a[j] = v;
    }
#pragma unroll
    for (int j = 0; j < 8; ++j) { b0[j] = wp0[j]; b0[j + 8] = wp0[16 + j]; b1[j] = wp1[j]; b1[j + 8] = wp1[16 + j]; }
    acc[0] = wmma_f16(a, b0, acc[0]);
    acc[1] = wmma_f16(a, b1, acc[1]);
    wp0 += 32; wp1 += 32;
  }
#pragma unroll
  for (int tile = 0; tile < 2; ++tile) {
    int n = n0 + tile * 16;
#pragma unroll
    for (int i = 0; i < 8; ++i) {
      int m = m0 + i + hi * 8;             // chunk-local row
      float v = acc[tile][i] * g1[n] + b1[n];
      v = fmaxf(v, 0.f);
      h1[(size_t)m * mid + n] = (_Float16)v;
    }
  }
}

// ---------------------------------------------------------------------------
// Stage GEMM2 fused with neighborhood max-pool + skip + relu.
// One block per (b,s) center: the 32 x mid h1 tile is staged in LDS once via
// the Tensor Data Mover (async DMA, TENSORcnt) when available; 8 waves then
// sweep the co column tiles with 2x2 WMMA register blocking; max over the 32
// neighbor rows = 16-elem register reduction + shfl_xor(16) across lane halves.
// ---------------------------------------------------------------------------
__global__ void k_gemm2_pool(const _Float16* __restrict__ h1, const _Float16* __restrict__ W2,
                             const float* __restrict__ g2, const float* __restrict__ b2,
                             const float* __restrict__ identity, _Float16* __restrict__ fout,
                             int bbase, int S, int mid, int co) {
  __shared__ _Float16 sh[32 * 256];
  int blk = blockIdx.x;                      // chunk-local (b,s)
  int bs_abs = (bbase + blk / S) * S + (blk % S);
  int rowbase = blk * 32;                    // chunk-local h1 row base

#if __has_builtin(__builtin_amdgcn_tensor_load_to_lds)
  if (threadIdx.x < 32) {                    // one wave issues the tensor DMA
    unsigned long long ga = (unsigned long long)(h1 + (size_t)rowbase * mid);
    unsigned lds_addr = (unsigned)(unsigned long long)(void*)&sh[0];
    unsigned nelem = 32u * (unsigned)mid;    // contiguous 2-byte elements
    tdm_v4u g0;
    g0[0] = 1u;                                              // count=1, user D#
    g0[1] = lds_addr;                                        // lds_addr
    g0[2] = (unsigned)(ga & 0xffffffffull);                  // global_addr[31:0]
    g0[3] = (unsigned)((ga >> 32) & 0x01ffffffull) | (2u << 30); // addr[56:32], type=2
    tdm_v8i g1;
    g1[0] = (int)(1u << 16);                                 // data_size = 2 bytes
    g1[1] = (int)((nelem & 0xffffu) << 16);                  // tensor_dim0[15:0]
    g1[2] = (int)((1u << 16) | (nelem >> 16));               // tensor_dim1=1 | dim0[31:16]
    g1[3] = (int)((nelem & 0xffffu) << 16);                  // tile_dim0
    g1[4] = 1;                                               // tile_dim1 = 1
    g1[5] = (int)nelem;                                      // tensor_dim0_stride
    g1[6] = 0;
    g1[7] = 0;
    tdm_v4i zg2 = {};
    tdm_v4i zg3 = {};
#if __clang_major__ >= 23
    tdm_v8i zg4 = {};
    __builtin_amdgcn_tensor_load_to_lds(g0, g1, zg2, zg3, zg4, 0);
#else
    __builtin_amdgcn_tensor_load_to_lds(g0, g1, zg2, zg3, 0);
#endif
    __builtin_amdgcn_s_wait_tensorcnt(0);
  }
#else
  for (int i = threadIdx.x; i < 32 * mid; i += blockDim.x)
    sh[i] = h1[(size_t)rowbase * mid + i];
#endif
  __syncthreads();

  int lane = threadIdx.x & 31;
  int wv   = threadIdx.x >> 5;
  int hi   = lane >> 4;
  int rlo  = (lane & 15);
  for (int ntp = wv; ntp * 32 < co; ntp += 8) {
    int n0 = ntp * 32 + (lane & 15);
    v8f acc[4] = {};   // [0]=t0 rows0-15 [1]=t0 rows16-31 [2]=t1 rows0-15 [3]=t1 rows16-31
    const _Float16* wp0 = W2 + (size_t)n0 * mid + hi * 8;
    const _Float16* wp1 = wp0 + (size_t)16 * mid;
    for (int k0 = 0; k0 < mid; k0 += 32) {
      v16h a0, a1, b0, b1;
#pragma unroll
      for (int j = 0; j < 16; ++j) {
        int kk = a_kidx(k0, hi, j);
        a0[j] = sh[(size_t)rlo * mid + kk];
        a1[j] = sh[(size_t)(16 + rlo) * mid + kk];
      }
#pragma unroll
      for (int j = 0; j < 8; ++j) { b0[j] = wp0[j]; b0[j + 8] = wp0[16 + j]; b1[j] = wp1[j]; b1[j + 8] = wp1[16 + j]; }
      acc[0] = wmma_f16(a0, b0, acc[0]);
      acc[1] = wmma_f16(a1, b0, acc[1]);
      acc[2] = wmma_f16(a0, b1, acc[2]);
      acc[3] = wmma_f16(a1, b1, acc[3]);
      wp0 += 32; wp1 += 32;
    }
#pragma unroll
    for (int tile = 0; tile < 2; ++tile) {
      float mv = acc[tile * 2][0];
#pragma unroll
      for (int i = 1; i < 8; ++i) mv = fmaxf(mv, acc[tile * 2][i]);
#pragma unroll
      for (int i = 0; i < 8; ++i) mv = fmaxf(mv, acc[tile * 2 + 1][i]);
      mv = fmaxf(mv, __shfl_xor(mv, 16, 32));   // combine the two M-halves
      if (lane < 16) {
        int n = ntp * 32 + tile * 16 + lane;
        float v = mv * g2[n] + b2[n] + identity[(size_t)bs_abs * co + n];
        v = fmaxf(v, 0.f);
        fout[(size_t)bs_abs * co + n] = (_Float16)v;
      }
    }
  }
}

// ---------------------------------------------------------------------------
// features = max over S rows of h (B,S,C) -> (B,C) fp32
// ---------------------------------------------------------------------------
__global__ void k_rowmax(const _Float16* __restrict__ h, float* __restrict__ out, int S, int C) {
  int t = blockIdx.x * blockDim.x + threadIdx.x;
  if (t >= B_ * C) return;
  int b = t / C, c = t % C;
  float m = -1e30f;
  for (int s = 0; s < S; ++s) m = fmaxf(m, (float)h[((size_t)b * S + s) * C + c]);
  out[(size_t)b * C + c] = m;
}

// SA input: [p(3), f(Cf), 0-pad] per (b,s) row -> f16 (B*S, Kp)
__global__ void k_build_sa(const float* __restrict__ p, const _Float16* __restrict__ f,
                           _Float16* __restrict__ A, int S, int Cf, int Kp) {
  int t = blockIdx.x * blockDim.x + threadIdx.x;
  int total = B_ * S * Kp;
  if (t >= total) return;
  int row = t / Kp, c = t % Kp;
  _Float16 v;
  if (c < 3)            v = (_Float16)p[(size_t)row * 3 + c];
  else if (c < 3 + Cf)  v = f[(size_t)row * Cf + (c - 3)];
  else                  v = (_Float16)0.f;
  A[t] = v;
}

// x = [features(512), info @ info_w.T + info_b (32)] -> f16 (B,544)
__global__ void k_build_x(const float* __restrict__ feats, const float* __restrict__ info,
                          const float* __restrict__ iw, const float* __restrict__ ib,
                          _Float16* __restrict__ x) {
  int t = blockIdx.x * blockDim.x + threadIdx.x;
  if (t >= B_ * 544) return;
  int b = t / 544, c = t % 544;
  float v;
  if (c < 512) v = feats[(size_t)b * 512 + c];
  else {
    int j = c - 512;
    v = ib[j];
    for (int k = 0; k < 3; ++k) v += iw[j * 3 + k] * info[b * 3 + k];
  }
  x[t] = (_Float16)v;
}

// Exact per-row LayerNorm + relu: fp32 (rows,C) -> f16
__global__ void k_ln_relu(const float* __restrict__ z, const float* __restrict__ g,
                          const float* __restrict__ be, _Float16* __restrict__ out, int C) {
  __shared__ float s1[256], s2[256];
  int r = blockIdx.x;
  const float* zr = z + (size_t)r * C;
  float a = 0.f, bb = 0.f;
  for (int c = threadIdx.x; c < C; c += blockDim.x) { float v = zr[c]; a += v; bb += v * v; }
  s1[threadIdx.x] = a; s2[threadIdx.x] = bb;
  __syncthreads();
  for (int o = blockDim.x >> 1; o > 0; o >>= 1) {
    if ((int)threadIdx.x < o) { s1[threadIdx.x] += s1[threadIdx.x + o]; s2[threadIdx.x] += s2[threadIdx.x + o]; }
    __syncthreads();
  }
  float mu = s1[0] / C;
  float var = s2[0] / C - mu * mu;
  float rs = rsqrtf(var + 1e-5f);
  for (int c = threadIdx.x; c < C; c += blockDim.x) {
    float v = (zr[c] - mu) * rs * g[c] + be[c];
    out[(size_t)r * C + c] = (_Float16)fmaxf(v, 0.f);
  }
}

// ---------------------------------------------------------------------------
// Host orchestration
// ---------------------------------------------------------------------------
extern "C" void kernel_launch(void* const* d_in, const int* in_sizes, int n_in,
                              void* d_out, int out_size, void* d_ws, size_t ws_size,
                              hipStream_t stream) {
  (void)in_sizes; (void)n_in; (void)out_size; (void)ws_size;

  const float* points = (const float*)d_in[0];
  const float* info   = (const float*)d_in[1];
  const float* head_w = (const float*)d_in[2];
  const float* head_b = (const float*)d_in[3];
  auto SP = [&](int s, int j) { return (const float*)d_in[4 + 8 * s + j]; };
  const float* sa_w1 = (const float*)d_in[36];
  const float* sa_g1 = (const float*)d_in[37];
  const float* sa_b1 = (const float*)d_in[38];
  const float* sa_w2 = (const float*)d_in[39];
  const float* sa_g2 = (const float*)d_in[40];
  const float* sa_b2 = (const float*)d_in[41];
  const float* info_w = (const float*)d_in[42];
  const float* info_b = (const float*)d_in[43];
  const float* hW1[2] = { (const float*)d_in[44], (const float*)d_in[50] };
  const float* hB1[2] = { (const float*)d_in[45], (const float*)d_in[51] };
  const float* hG [2] = { (const float*)d_in[46], (const float*)d_in[52] };
  const float* hBe[2] = { (const float*)d_in[47], (const float*)d_in[53] };
  const float* hW2[2] = { (const float*)d_in[48], (const float*)d_in[54] };
  const float* hB2[2] = { (const float*)d_in[49], (const float*)d_in[55] };

  const int ci[4]    = { 32, 64, 128, 256 };
  const int co[4]    = { 64, 128, 256, 512 };
  const int mid[4]   = { 32, 64, 128, 256 };
  const int cin1[4]  = { 35, 67, 131, 259 };
  const int cinp[4]  = { 64, 96, 160, 288 };
  const int Nprev[4] = { 1024, 512, 256, 128 };
  const float radii[4] = { 0.15f, 0.225f, 0.3375f, 0.50625f };

  // ---- workspace bump allocator
  char* wsp = (char*)d_ws;
  size_t off = 0;
  auto alloc = [&](size_t bytes) -> void* {
    void* p = wsp + off;
    off = (off + bytes + 255) & ~(size_t)255;
    return p;
  };
  auto allocH = [&](size_t n) { return (_Float16*)alloc(n * sizeof(_Float16)); };
  auto allocF = [&](size_t n) { return (float*)alloc(n * sizeof(float)); };
  auto allocI = [&](size_t n) { return (int*)alloc(n * sizeof(int)); };

  // packed f16 weights
  _Float16 *w1p[4], *w2p[4], *skp[4];
  for (int s = 0; s < 4; ++s) {
    w1p[s] = allocH((size_t)mid[s] * cinp[s]);
    w2p[s] = allocH((size_t)co[s] * mid[s]);
    skp[s] = allocH((size_t)co[s] * ci[s]);
  }
  _Float16* sa1p = allocH((size_t)512 * 544);
  _Float16* sa2p = allocH((size_t)512 * 512);
  _Float16* hw1p[2] = { allocH((size_t)256 * 544), allocH((size_t)256 * 544) };
  _Float16* hw2p[2] = { allocH((size_t)64 * 256),  allocH((size_t)160 * 256) };

  // data buffers
  float*    pb[5]; _Float16* fb[5];
  int Npts[5] = { 1024, 512, 256, 128, 64 };
  int fch [5] = { 32, 64, 128, 256, 512 };
  for (int i = 0; i < 5; ++i) {
    pb[i] = allocF((size_t)B_ * Npts[i] * 3);
    fb[i] = allocH((size_t)B_ * Npts[i] * fch[i]);
  }
  int*      idxb  = allocI((size_t)B_ * 512);
  int*      gidxb = allocI((size_t)B_ * 512 * NSAMP_);
  _Float16* fib   = allocH((size_t)B_ * 512 * 32);          // max B*S*ci
  float*    identb= allocF((size_t)B_ * 512 * 64);          // max B*S*co (==B*64*512)
  _Float16* h1b   = allocH((size_t)CHUNK_B * 512 * 32 * 32);// constant 8.4M halves/stage
  _Float16* Asab  = allocH((size_t)B_ * 64 * 544);
  _Float16* hsa1  = allocH((size_t)B_ * 64 * 512);
  _Float16* hsa2  = allocH((size_t)B_ * 64 * 512);
  _Float16* xb    = allocH((size_t)B_ * 544);
  float*    zb    = allocF((size_t)B_ * 256);
  _Float16* zhb   = allocH((size_t)B_ * 256);

  float* featsOut = (float*)d_out;                  // (128,512)
  float* predOut  = featsOut + (size_t)B_ * 512;    // (128,49)
  float* offOut   = predOut  + (size_t)B_ * 49;     // (128,147)

  auto pack = [&](const float* src, _Float16* dst, int r, int c, int rp, int cp) {
    int total = rp * cp;
    k_pack_w<<<(total + 255) / 256, 256, 0, stream>>>(src, dst, r, c, rp, cp);
  };

  // ---- pack all weights
  for (int s = 0; s < 4; ++s) {
    pack(SP(s, 0), w1p[s], mid[s], cin1[s], mid[s], cinp[s]);
    pack(SP(s, 3), w2p[s], co[s],  mid[s],  co[s],  mid[s]);
    pack(SP(s, 6), skp[s], co[s],  ci[s],   co[s],  ci[s]);
  }
  pack(sa_w1, sa1p, 512, 515, 512, 544);
  pack(sa_w2, sa2p, 512, 512, 512, 512);
  for (int h = 0; h < 2; ++h) pack(hW1[h], hw1p[h], 256, 544, 256, 544);
  pack(hW2[0], hw2p[0], 49, 256, 64, 256);
  pack(hW2[1], hw2p[1], 147, 256, 160, 256);

  // ---- head
  {
    int total = B_ * 1024;
    k_head<<<(total + 255) / 256, 256, 0, stream>>>(points, head_w, head_b, pb[0], fb[0], 1024);
  }

  // ---- 4 set-abstraction stages
  for (int s = 0; s < 4; ++s) {
    int N = Nprev[s], S = N / 2;
    float r = radii[s], r2 = r * r;

    k_fps<<<B_, 256, 0, stream>>>(pb[s], N, S, idxb);

    {
      int total = B_ * S;
      k_gather<<<(total + 255) / 256, 256, 0, stream>>>(pb[s], fb[s], idxb, pb[s + 1], fib, N, S, ci[s]);
    }

    // identity = fi @ skip_w.T + skip_b   (fp32, (B*S, co))
    {
      dim3 g((B_ * S) / 16, (co[s] / 32 + 7) / 8);
      k_gemm<<<g, 256, 0, stream>>>(fib, skp[s], SP(s, 7), nullptr, nullptr,
                                    identb, nullptr, B_ * S, ci[s], co[s], co[s], co[s], 0);
    }

    {
      int total = B_ * S;
      k_ball<<<(total + 255) / 256, 256, 0, stream>>>(pb[s], pb[s + 1], gidxb, N, S, r2);
    }

    // chunked fused MLP1 -> MLP2+pool (h1 stays L2-resident)
    for (int cb = 0; cb < B_ / CHUNK_B; ++cb) {
      int bbase = cb * CHUNK_B;
      int Mch = CHUNK_B * S * 32;
      dim3 g1(Mch / 16, (mid[s] / 32 + 7) / 8);
      k_gemm1_feat<<<g1, 256, 0, stream>>>(pb[s], fb[s], pb[s + 1], gidxb, w1p[s],
                                           SP(s, 1), SP(s, 2), h1b,
                                           bbase, S, N, ci[s], cinp[s], mid[s], 1.f / r);
      k_gemm2_pool<<<CHUNK_B * S, 256, 0, stream>>>(h1b, w2p[s], SP(s, 4), SP(s, 5),
                                                    identb, fb[s + 1], bbase, S, mid[s], co[s]);
    }
  }

  // ---- SA global MLPs
  {
    int total = B_ * 64 * 544;
    k_build_sa<<<(total + 255) / 256, 256, 0, stream>>>(pb[4], fb[4], Asab, 64, 512, 544);
  }
  {
    dim3 g((B_ * 64) / 16, (512 / 32 + 7) / 8);
    k_gemm<<<g, 256, 0, stream>>>(Asab, sa1p, nullptr, sa_g1, sa_b1,
                                  nullptr, hsa1, B_ * 64, 544, 512, 512, 512, 1);
    k_gemm<<<g, 256, 0, stream>>>(hsa1, sa2p, nullptr, sa_g2, sa_b2,
                                  nullptr, hsa2, B_ * 64, 512, 512, 512, 512, 1);
  }
  {
    int total = B_ * 512;
    k_rowmax<<<(total + 255) / 256, 256, 0, stream>>>(hsa2, featsOut, 64, 512);
  }
  {
    int total = B_ * 544;
    k_build_x<<<(total + 255) / 256, 256, 0, stream>>>(featsOut, info, info_w, info_b, xb);
  }

  // ---- heads (anchor: 49 cols, offset: 147 cols)
  const int Npadh[2] = { 64, 160 };
  const int Nacth[2] = { 49, 147 };
  float* outPtr[2] = { predOut, offOut };
  for (int h = 0; h < 2; ++h) {
    dim3 g1(B_ / 16, (256 / 32 + 7) / 8);
    k_gemm<<<g1, 256, 0, stream>>>(xb, hw1p[h], hB1[h], nullptr, nullptr,
                                   zb, nullptr, B_, 544, 256, 256, 256, 0);
    k_ln_relu<<<B_, 256, 0, stream>>>(zb, hG[h], hBe[h], zhb, 256);
    dim3 g2(B_ / 16, (Npadh[h] / 32 + 7) / 8);
    k_gemm<<<g2, 256, 0, stream>>>(zhb, hw2p[h], hB2[h], nullptr, nullptr,
                                   outPtr[h], nullptr, B_, 256, Npadh[h], Nacth[h], Nacth[h], 0);
  }
}
